// HolE_44470091383206
// MI455X (gfx1250) — matrix operand order repaired and checked by reading, not compiled, not run on hardware
//
#include <hip/hip_runtime.h>
#include <hip/hip_bf16.h>
#include <math.h>

// CDNA5 WMMA vector types
typedef _Float16 v16h __attribute__((ext_vector_type(16)));
typedef float    v8f  __attribute__((ext_vector_type(8)));

#define KD 256
#define WAVES_PER_BLOCK 8

// Per-wave LDS staging. Offsets chosen so every vector access is 16B aligned:
//   s  at +0    (1024 B, f32)
//   p  at +1024 (512 B, f16)
//   o0 at +1536 (1024 B, f16, o duplicated: o0[i] = o[i % 256])
//   o1 at +2560 (1024 B, f16, shifted:     o1[i] = o[(i+1) % 256])
struct __align__(16) WaveLDS {
  float    s[KD];
  _Float16 p[KD];
  _Float16 o0[2 * KD];
  _Float16 o1[2 * KD];
};

__global__ __launch_bounds__(32 * WAVES_PER_BLOCK)
void hole_wmma_kernel(const int* __restrict__ triples,
                      const float* __restrict__ ent_emb,
                      const float* __restrict__ rel_emb,
                      const float* __restrict__ gamma,
                      const float* __restrict__ beta,
                      const float* __restrict__ mmean,
                      const float* __restrict__ mvar,
                      float* __restrict__ out,
                      int batch)
{
  __shared__ WaveLDS lds[WAVES_PER_BLOCK];

  const int wave = threadIdx.x >> 5;
  const int lane = threadIdx.x & 31;
  int t = blockIdx.x * WAVES_PER_BLOCK + wave;
  const bool valid = (t < batch);
  if (!valid) t = batch - 1;   // clamp: keep whole block converged for barriers/WMMA

  WaveLDS& W = lds[wave];

  // ---- gather indices (broadcast loads) ----
  const int sidx = triples[3 * t + 0];
  const int pidx = triples[3 * t + 1];
  const int oidx = triples[3 * t + 2];

  const float4* srow = (const float4*)(ent_emb + (size_t)sidx * KD);
  const float4* prow = (const float4*)(rel_emb + (size_t)pidx * KD);
  const float4* orow = (const float4*)(ent_emb + (size_t)oidx * KD);

  // ---- stage s (f32) ----
  float4 sA = srow[2 * lane];
  float4 sB = srow[2 * lane + 1];
  ((float4*)W.s)[2 * lane]     = sA;
  ((float4*)W.s)[2 * lane + 1] = sB;

  union H8 { _Float16 h[8]; uint4 q; };

  // ---- stage p (f16) ----
  {
    float4 a = prow[2 * lane], b = prow[2 * lane + 1];
    H8 ph;
    ph.h[0] = (_Float16)a.x; ph.h[1] = (_Float16)a.y;
    ph.h[2] = (_Float16)a.z; ph.h[3] = (_Float16)a.w;
    ph.h[4] = (_Float16)b.x; ph.h[5] = (_Float16)b.y;
    ph.h[6] = (_Float16)b.z; ph.h[7] = (_Float16)b.w;
    *(uint4*)&W.p[8 * lane] = ph.q;
  }

  // ---- stage o (f16): duplicated copy + one-element-shifted duplicated copy ----
  {
    float4 a = orow[2 * lane], b = orow[2 * lane + 1];
    H8 oh;
    oh.h[0] = (_Float16)a.x; oh.h[1] = (_Float16)a.y;
    oh.h[2] = (_Float16)a.z; oh.h[3] = (_Float16)a.w;
    oh.h[4] = (_Float16)b.x; oh.h[5] = (_Float16)b.y;
    oh.h[6] = (_Float16)b.z; oh.h[7] = (_Float16)b.w;
    *(uint4*)&W.o0[8 * lane]      = oh.q;
    *(uint4*)&W.o0[8 * lane + KD] = oh.q;
    #pragma unroll
    for (int u = 0; u < 8; ++u) {
      int j = 8 * lane + u;
      _Float16 hv = oh.h[u];
      W.o1[(j + 511) & 511] = hv;   // o1[i] = o[(i+1)%256], low copy
      W.o1[(j + 255) & 511] = hv;   // high copy
    }
  }

  __syncthreads();

  // ---- 8x v_wmma_f32_16x16x32_f16: C = sum_H [B_H | B_{H+1}] x [P^(H); P^(H+1)] ----
  v8f c = {};
  const int m  = lane & 15;            // matrix row (A) / column (B) for this lane
  const int lo = (lane < 16) ? 1 : 0;

  #pragma unroll
  for (int Hh = 0; Hh < 8; ++Hh) {
    const int H = 2 * Hh;

    // A operand: Hankel rows of o. Lane needs o[e..e+15],
    // e = 16H + m + (lane<16 ? 0 : 16). Parity-select copy so loads are 4B aligned.
    union { unsigned int u[8]; v16h h; } av;
    const int e = 16 * H + m + (lo ? 0 : 16);
    const unsigned int* ap = (e & 1)
        ? (const unsigned int*)&W.o1[e - 1]
        : (const unsigned int*)&W.o0[e];
    av.u[0] = ap[0];  av.u[1] = ap[1];  av.u[2] = ap[2];  av.u[3] = ap[3];
    av.u[4] = ap[4];  av.u[5] = ap[5];  av.u[6] = ap[6];  av.u[7] = ap[7];

    // B operand: permuted p-blocks. Lane (col n=m) needs the aligned 16-half block
    // p[16*((H' - n) & 15) .. +15], H' = H (lanes 0-15) or H+1 (lanes 16-31).
    union { uint4 q[2]; v16h h; } bv;
    const int blk = (((lo ? H : H + 1) - m) & 15);
    const uint4* bp = (const uint4*)&W.p[16 * blk];
    bv.q[0] = bp[0];
    bv.q[1] = bp[1];

    c = __builtin_amdgcn_wmma_f32_16x16x32_f16(
        /*neg_a=*/false, av.h, /*neg_b=*/false, bv.h,
        /*c_mod=*/(short)0, c, /*reuse_a=*/false, /*reuse_b=*/false);
  }

  // ---- score = <S, C> using documented C/D layout (VGPR r: M=r / r+8; lane: N) ----
  const float* srd = lo ? &W.s[16 * m] : &W.s[16 * m + 8];
  float4 s0 = ((const float4*)srd)[0];
  float4 s1 = ((const float4*)srd)[1];
  float partial = s0.x * c[0] + s0.y * c[1] + s0.z * c[2] + s0.w * c[3]
                + s1.x * c[4] + s1.y * c[5] + s1.z * c[6] + s1.w * c[7];

  #pragma unroll
  for (int off = 16; off > 0; off >>= 1)
    partial += __shfl_xor(partial, off, 32);

  if (valid && lane == 0) {
    float sig = 1.0f / (1.0f + __expf(-partial));
    float inv = rsqrtf(mvar[0] + 0.001f);
    out[t] = (sig - mmean[0]) * inv * gamma[0] + beta[0];
  }
}

extern "C" void kernel_launch(void* const* d_in, const int* in_sizes, int n_in,
                              void* d_out, int out_size, void* d_ws, size_t ws_size,
                              hipStream_t stream) {
  const int*   triples = (const int*)  d_in[0];
  const float* ent_emb = (const float*)d_in[1];
  const float* rel_emb = (const float*)d_in[2];
  const float* gamma   = (const float*)d_in[3];
  const float* beta    = (const float*)d_in[4];
  const float* mmean   = (const float*)d_in[5];
  const float* mvar    = (const float*)d_in[6];
  float* out = (float*)d_out;

  const int batch  = in_sizes[0] / 3;
  const int blocks = (batch + WAVES_PER_BLOCK - 1) / WAVES_PER_BLOCK;

  hole_wmma_kernel<<<blocks, 32 * WAVES_PER_BLOCK, 0, stream>>>(
      triples, ent_emb, rel_emb, gamma, beta, mmean, mvar, out, batch);
}
